// GraphConv_and_gather_60086592471829
// MI455X (gfx1250) — matrix operand order repaired and checked by reading, not compiled, not run on hardware
//
#include <hip/hip_runtime.h>
#include <math.h>

typedef __attribute__((ext_vector_type(16))) _Float16 v16h;
typedef __attribute__((ext_vector_type(8)))  float    v8f;

#define MAX_DEG 10
#define PER_DEG 50000
#define N_ATOMS (PER_DEG * (MAX_DEG + 1))   // 550000
#define BATCH   4096
#define F       128
#define EPS     1e-5f
#define NTILES  (N_ATOMS / 16)              // 34375 (PER_DEG % 16 == 0)

struct AdjPtrs { const int* a[MAX_DEG]; };

union Frag16 { v16h v; float4 f4[2]; };
struct h4 { _Float16 x, y, z, w; };         // 8-byte packed half quad

// ---------------------------------------------------------------------------
// Pack the 21 W matrices + 11 gW matrices (128x128 f32, row-major [k][n]) into
// f16 B-fragment order for v_wmma_f32_16x16x32_f16:
//   entry(mat, jtile, kblk) = 32 lanes x 16 halves (contiguous per lane)
//   lane l: column n = 16*jtile + (l&15); k = 32*kblk + ((l<16)?0:16) + half
// ---------------------------------------------------------------------------
__global__ void prep_pack_w(const float* __restrict__ W,
                            const float* __restrict__ gW,
                            _Float16* __restrict__ packed) {
    int tid = blockIdx.x * blockDim.x + threadIdx.x;   // 32*8*4*32*16 = 524288
    int h = tid & 15;
    int l = (tid >> 4) & 31;
    int t = (tid >> 9) & 3;
    int j = (tid >> 11) & 7;
    int m = tid >> 14;                                  // 0..31
    const float* src = (m < 21) ? (W + (size_t)m * F * F)
                                : (gW + (size_t)(m - 21) * F * F);
    int n = 16 * j + (l & 15);
    int k = 32 * t + ((l < 16) ? 0 : 16) + h;
    packed[tid] = (_Float16)src[k * F + n];
}

__global__ void zero_gfeat(float* __restrict__ gfeat) {
    int tid = blockIdx.x * blockDim.x + threadIdx.x;
    if (tid < BATCH * F) gfeat[tid] = 0.f;
}

// ---------------------------------------------------------------------------
// Main fused kernel: one 16-row atom tile per block (256 thr = 8 waves).
// ---------------------------------------------------------------------------
__launch_bounds__(256)
__global__ void graphconv_main(const float* __restrict__ atoms,
                               const int*   __restrict__ membership,
                               AdjPtrs adj,
                               const _Float16* __restrict__ packedW,
                               const float* __restrict__ b,
                               const float* __restrict__ gb,
                               const float* __restrict__ gamma,
                               const float* __restrict__ beta,
                               float* __restrict__ out_normed,
                               float* __restrict__ gfeat) {
    __shared__ alignas(16) _Float16 sSelf[16][136];   // row stride 272B (16B aligned)
    __shared__ alignas(16) _Float16 sNbr[16][136];
    __shared__ alignas(16) float sRel[16][132];       // padded f32 rel tile
    __shared__ float sMean[16];
    __shared__ float sInv[16];

    const int tile = blockIdx.x;
    const int row0 = tile * 16;
    const int d    = row0 / PER_DEG;                  // uniform per block
    const int tid  = threadIdx.x;

    // ---- phase 1: stage self rows + neighbor sums (b128 gathers -> f16 LDS) ----
    const int* adjp = (d > 0) ? adj.a[d - 1] : nullptr;
    const int  lr   = row0 - d * PER_DEG;
    // 16 rows x 32 float4 = 512 vec4 slots; each of 256 threads does 2.
    for (int idx = tid; idx < 16 * (F / 4); idx += 256) {
        int r = idx >> 5, c4 = idx & 31;              // wave covers a full row
        const float4* srow = (const float4*)(atoms + (size_t)(row0 + r) * F);
        float4 sv = srow[c4];
        h4 hs; hs.x = (_Float16)sv.x; hs.y = (_Float16)sv.y;
               hs.z = (_Float16)sv.z; hs.w = (_Float16)sv.w;
        *(h4*)(&sSelf[r][c4 * 4]) = hs;
        if (d > 0) {
            float ax = 0.f, ay = 0.f, az = 0.f, aw = 0.f;
            for (int i = 0; i < d; ++i) {
                int nb = adjp[(lr + r) * d + i];      // broadcast across row's lanes
                const float4* nrow = (const float4*)(atoms + (size_t)nb * F);
                float4 nv = nrow[c4];                 // global_load_b128
                ax += nv.x; ay += nv.y; az += nv.z; aw += nv.w;
            }
            h4 hn; hn.x = (_Float16)ax; hn.y = (_Float16)ay;
                   hn.z = (_Float16)az; hn.w = (_Float16)aw;
            *(h4*)(&sNbr[r][c4 * 4]) = hn;
        }
    }
    __syncthreads();

    // ---- phase 2: WMMA GEMMs, one 16-col slice per wave ----
    const int wave  = tid >> 5;
    const int lane  = tid & 31;
    const int n     = lane & 15;
    const int col   = wave * 16 + n;
    const int kbase = (lane < 16) ? 0 : 16;           // A-fragment K split
    const int m_lo  = (lane < 16) ? 0 : 8;            // C/D row split

    int wrel0, wrel1, gwi;
    if (d == 0) { wrel0 = -1; wrel1 = 2 * MAX_DEG; gwi = MAX_DEG; }
    else        { wrel0 = 2 * (d - 1); wrel1 = wrel0 + 1; gwi = d - 1; }

    const float biasRel = b[wrel1 * F + col] + ((d > 0) ? b[wrel0 * F + col] : 0.f);
    const float biasGat = gb[gwi * F + col];

    v8f accR, accG;
#pragma unroll
    for (int i = 0; i < 8; ++i) { accR[i] = biasRel; accG[i] = biasGat; }

    const int j = wave;
#pragma unroll
    for (int t = 0; t < 4; ++t) {
        // A fragment (self): lane holds row n, K = {k0..k0+15}
        const float4* ps = (const float4*)(&sSelf[n][32 * t + kbase]);
        Frag16 aS; aS.f4[0] = ps[0]; aS.f4[1] = ps[1];

        // B fragment for self-rel weight W[wrel1]
        {
            const float4* pb = (const float4*)(packedW +
                (size_t)(((wrel1 * 8 + j) * 4 + t)) * 512 + lane * 16);
            Frag16 bf; bf.f4[0] = pb[0]; bf.f4[1] = pb[1];
            accR = __builtin_amdgcn_wmma_f32_16x16x32_f16(
                       false, aS.v, false, bf.v, (short)0, accR, false, false);
        }
        // B fragment for gather weight gW[gwi] (packed index 21+gwi)
        {
            const float4* pb = (const float4*)(packedW +
                (size_t)((((21 + gwi) * 8 + j) * 4 + t)) * 512 + lane * 16);
            Frag16 bf; bf.f4[0] = pb[0]; bf.f4[1] = pb[1];
            accG = __builtin_amdgcn_wmma_f32_16x16x32_f16(
                       false, aS.v, false, bf.v, (short)0, accG, false, false);
        }
        if (d > 0) {   // uniform branch; EXEC stays all-ones
            const float4* pn = (const float4*)(&sNbr[n][32 * t + kbase]);
            Frag16 aN; aN.f4[0] = pn[0]; aN.f4[1] = pn[1];
            const float4* pb = (const float4*)(packedW +
                (size_t)(((wrel0 * 8 + j) * 4 + t)) * 512 + lane * 16);
            Frag16 bf; bf.f4[0] = pb[0]; bf.f4[1] = pb[1];
            accR = __builtin_amdgcn_wmma_f32_16x16x32_f16(
                       false, aN.v, false, bf.v, (short)0, accR, false, false);
        }
    }

    // rel tile -> LDS (C/D layout: element (m_lo + r, n))
#pragma unroll
    for (int r = 0; r < 8; ++r) sRel[m_lo + r][col] = accR[r];

    // gather tile -> segment-sum via f32 global atomics
#pragma unroll
    for (int r = 0; r < 8; ++r) {
        int mm  = m_lo + r;
        int seg = membership[row0 + mm];
        __hip_atomic_fetch_add(&gfeat[(size_t)seg * F + col], accG[r],
                               __ATOMIC_RELAXED, __HIP_MEMORY_SCOPE_AGENT);
    }
    __syncthreads();

    // ---- phase 3: row-wise layernorm of rel tile ----
    if (tid < 16) {
        float s = 0.f;
        for (int c = 0; c < F; ++c) s += sRel[tid][c];
        float mean = s * (1.f / F);
        float v = 0.f;
        for (int c = 0; c < F; ++c) { float x = sRel[tid][c] - mean; v += x * x; }
        float stdv = sqrtf(v * (1.f / F) + EPS);
        sMean[tid] = mean;
        sInv[tid]  = 1.f / (stdv + EPS);
    }
    __syncthreads();

    // vectorized normed-output store (global_store_b128)
    for (int idx = tid; idx < 16 * (F / 4); idx += 256) {
        int r = idx >> 5, c4 = idx & 31;
        float mean = sMean[r], inv = sInv[r];
        const float4 g = ((const float4*)gamma)[c4];
        const float4 be = ((const float4*)beta)[c4];
        float4 o;
        o.x = g.x * (sRel[r][c4 * 4 + 0] - mean) * inv + be.x;
        o.y = g.y * (sRel[r][c4 * 4 + 1] - mean) * inv + be.y;
        o.z = g.z * (sRel[r][c4 * 4 + 2] - mean) * inv + be.z;
        o.w = g.w * (sRel[r][c4 * 4 + 3] - mean) * inv + be.w;
        ((float4*)(out_normed + (size_t)(row0 + r) * F))[c4] = o;
    }
}

// ---------------------------------------------------------------------------
extern "C" void kernel_launch(void* const* d_in, const int* in_sizes, int n_in,
                              void* d_out, int out_size, void* d_ws, size_t ws_size,
                              hipStream_t stream) {
    (void)in_sizes; (void)n_in; (void)out_size; (void)ws_size;

    const float* atoms      = (const float*)d_in[0];
    // d_in[1] = deg_slice (layout is implied by PER_DEG; unused)
    const int*   membership = (const int*)d_in[2];
    AdjPtrs adj;
    for (int i = 0; i < MAX_DEG; ++i) adj.a[i] = (const int*)d_in[3 + i];
    const float* W     = (const float*)d_in[13];
    const float* b     = (const float*)d_in[14];
    const float* gW    = (const float*)d_in[15];
    const float* gb    = (const float*)d_in[16];
    const float* gamma = (const float*)d_in[17];
    const float* beta  = (const float*)d_in[18];

    float* out_normed = (float*)d_out;
    float* gfeat      = out_normed + (size_t)N_ATOMS * F;   // output #2

    _Float16* packedW = (_Float16*)d_ws;   // 32 * 128 * 128 * 2B = 1 MB

    prep_pack_w<<<2048, 256, 0, stream>>>(W, gW, packedW);          // 524288 thr
    zero_gfeat<<<(BATCH * F + 255) / 256, 256, 0, stream>>>(gfeat);
    graphconv_main<<<NTILES, 256, 0, stream>>>(atoms, membership, adj, packedW,
                                               b, gb, gamma, beta,
                                               out_normed, gfeat);
}
